// NA3DBlock_80195629351483
// MI455X (gfx1250) — matrix operand order repaired and checked by reading, not compiled, not run on hardware
//
#include <hip/hip_runtime.h>
#include <hip/hip_bf16.h>

// ---------------------------------------------------------------------------
// CDNA5 (gfx1250) implementation of a 3D neighborhood-attention block.
// All GEMMs + attention use v_wmma_f32_16x16x32_f16 (wave32 WMMA).
// ---------------------------------------------------------------------------

typedef __attribute__((ext_vector_type(16))) _Float16 v16h;
typedef __attribute__((ext_vector_type(8)))  float    v8f;

union HFrag {
    v16h     v;
    _Float16 h[16];
    uint4    q[2];
};

#define C_DIM   192
#define NVOX    4096
#define HID     384
#define NQKV    576
#define MLPH    768
#define NH      6
#define HD      32
#define KEYS    400   // 5*5*16 key tile per (head,d,h) query row
#define KPAD    416   // padded to multiple of 32 for the PV K-loop

// ---------------------------------------------------------------------------
// float -> f16 weight conversion
// ---------------------------------------------------------------------------
__global__ __launch_bounds__(256)
void f2h_kernel(const float* __restrict__ in, _Float16* __restrict__ out, int n) {
    int i = blockIdx.x * 256 + threadIdx.x;
    if (i < n) out[i] = (_Float16)in[i];
}

// ---------------------------------------------------------------------------
// per-channel stats: mean and 1/sqrt(mean(x^2)+eps) over the 4096 voxels
// ---------------------------------------------------------------------------
__global__ __launch_bounds__(256)
void chan_stats(const float* __restrict__ x, float* __restrict__ stats,
                float* __restrict__ rinv) {
    __shared__ float ss[256], sq[256];
    const int c = blockIdx.x;
    const float* p = x + (size_t)c * NVOX;
    float s = 0.f, q = 0.f;
    for (int i = threadIdx.x; i < NVOX; i += 256) { float v = p[i]; s += v; q += v * v; }
    ss[threadIdx.x] = s; sq[threadIdx.x] = q;
    __syncthreads();
    for (int o = 128; o > 0; o >>= 1) {
        if (threadIdx.x < o) {
            ss[threadIdx.x] += ss[threadIdx.x + o];
            sq[threadIdx.x] += sq[threadIdx.x + o];
        }
        __syncthreads();
    }
    if (threadIdx.x == 0) {
        stats[c] = ss[0] * (1.f / NVOX);
        rinv[c]  = rsqrtf(sq[0] * (1.f / NVOX) + 1e-6f);
    }
}

// ---------------------------------------------------------------------------
// gate MLP: h = relu(stats@w1^T+b1); g = sigmoid(h@w2^T+b2);
// coef[c] = rinv[c]*scale[c]*g[c]
// ---------------------------------------------------------------------------
__global__ __launch_bounds__(HID)
void gate_kernel(const float* __restrict__ stats, const float* __restrict__ rinv,
                 const float* __restrict__ w1, const float* __restrict__ b1,
                 const float* __restrict__ w2, const float* __restrict__ b2,
                 const float* __restrict__ scale, float* __restrict__ coef) {
    __shared__ float hbuf[HID];
    const int t = threadIdx.x;
    {
        float a = b1[t];
        const float* wr = w1 + (size_t)t * C_DIM;
        for (int c = 0; c < C_DIM; ++c) a += stats[c] * wr[c];
        hbuf[t] = fmaxf(a, 0.f);
    }
    __syncthreads();
    if (t < C_DIM) {
        float a = b2[t];
        const float* wr = w2 + (size_t)t * HID;
        for (int j = 0; j < HID; ++j) a += hbuf[j] * wr[j];
        float g = 1.f / (1.f + __expf(-a));
        coef[t] = rinv[t] * scale[t] * g;
    }
}

// ---------------------------------------------------------------------------
// normalize + transpose (C,vox) f32 -> (vox,C) f16
// ---------------------------------------------------------------------------
__global__ __launch_bounds__(256)
void normalize_t(const float* __restrict__ x, const float* __restrict__ coef,
                 _Float16* __restrict__ out) {
    int i = blockIdx.x * 256 + threadIdx.x;
    if (i >= C_DIM * NVOX) return;
    int c = i % C_DIM, m = i / C_DIM;
    out[i] = (_Float16)(x[(size_t)c * NVOX + m] * coef[c]);
}

// ---------------------------------------------------------------------------
// WMMA GEMM, register-blocked: each wave computes a 64x16 output tile
// (4 M-subtiles sharing every B fragment -> 4 WMMAs per weight load).
// K is compile-time so the K-loop fully unrolls and loads clause ahead.
// A: MxK f16 row-major, W: NxK f16 row-major.
// EPI 0: store f16 row-major      (qkv)
// EPI 1: exact GELU, store f16    (mlp1)
// EPI 2: +resid, store f32 channel-major [n*M+m]  (proj / mlp2)
// ---------------------------------------------------------------------------
template<int EPI, int K>
__global__ __launch_bounds__(256)
void gemm_wmma(const _Float16* __restrict__ A, const _Float16* __restrict__ W,
               const float* __restrict__ bias, int M, int N,
               _Float16* __restrict__ outH, float* __restrict__ outF,
               const float* __restrict__ resid) {
    const int lane = threadIdx.x & 31;
    const int wv   = threadIdx.x >> 5;
    const int ntn  = N >> 4;
    const int t    = blockIdx.x * 8 + wv;
    if (t >= (M >> 6) * ntn) return;           // wave-uniform exit (EXEC stays all-1)
    const int mt = t / ntn, nt = t % ntn;
    const int lm = lane & 15;
    const int hi = lane >> 4;
    const int m0 = mt << 6;

    const _Float16* arow0 = A + (size_t)(m0 + lm) * K;
    const _Float16* arow1 = arow0 + (size_t)16 * K;
    const _Float16* arow2 = arow0 + (size_t)32 * K;
    const _Float16* arow3 = arow0 + (size_t)48 * K;
    const _Float16* wrow  = W + (size_t)((nt << 4) + lm) * K;

    v8f c0 = {}, c1 = {}, c2 = {}, c3 = {};
#pragma unroll
    for (int k0 = 0; k0 < K; k0 += 32) {
        const int ka = k0 + (hi << 4);          // A 16x32 f16 lane layout
        const int kb = k0 + (hi << 4);          // B 32x16 f16 lane layout
        HFrag b, a0, a1, a2, a3;
        b.q[0]  = *(const uint4*)(wrow + kb);
        b.q[1]  = *(const uint4*)(wrow + kb + 8);
        a0.q[0] = *(const uint4*)(arow0 + ka);
        a0.q[1] = *(const uint4*)(arow0 + ka + 8);
        a1.q[0] = *(const uint4*)(arow1 + ka);
        a1.q[1] = *(const uint4*)(arow1 + ka + 8);
        a2.q[0] = *(const uint4*)(arow2 + ka);
        a2.q[1] = *(const uint4*)(arow2 + ka + 8);
        a3.q[0] = *(const uint4*)(arow3 + ka);
        a3.q[1] = *(const uint4*)(arow3 + ka + 8);
        c0 = __builtin_amdgcn_wmma_f32_16x16x32_f16(false, a0.v, false, b.v,
                                                    (short)0, c0, false, false);
        c1 = __builtin_amdgcn_wmma_f32_16x16x32_f16(false, a1.v, false, b.v,
                                                    (short)0, c1, false, false);
        c2 = __builtin_amdgcn_wmma_f32_16x16x32_f16(false, a2.v, false, b.v,
                                                    (short)0, c2, false, false);
        c3 = __builtin_amdgcn_wmma_f32_16x16x32_f16(false, a3.v, false, b.v,
                                                    (short)0, c3, false, false);
    }

    const int n  = (nt << 4) + lm;
    const float bn = bias[n];
    v8f acc[4] = {c0, c1, c2, c3};
#pragma unroll
    for (int s = 0; s < 4; ++s) {
#pragma unroll
        for (int r = 0; r < 8; ++r) {
            const int m = m0 + (s << 4) + r + (hi << 3);
            float v = acc[s][r] + bn;
            if (EPI == 0) {
                outH[(size_t)m * N + n] = (_Float16)v;
            } else if (EPI == 1) {
                float g = 0.5f * v * (1.f + erff(v * 0.70710678118654752f));
                outH[(size_t)m * N + n] = (_Float16)g;
            } else {
                size_t idx = (size_t)n * M + m;      // channel-major
                outF[idx] = resid[idx] + v;
            }
        }
    }
}

// ---------------------------------------------------------------------------
// Neighborhood attention. One wave per (head, d, h): 16 queries (w=0..15)
// against a shared 5x5x16 key tile (400 keys, padded to 416).
// Key fact: key index t = j*16 + lm has W coordinate (t&15) == lm, so the
// per-j part of every address is wave-uniform (zi=j/5, yi=j%5).
// qkv layout: [vox][576] f16 (q:0..191, k:192..383, v:384..575).
// obuf layout: [vox][192] f16.
// ---------------------------------------------------------------------------
__global__ __launch_bounds__(32)
void na3d_attn(const _Float16* __restrict__ qkv, _Float16* __restrict__ obuf) {
    __shared__ __attribute__((aligned(16))) _Float16 Vt[HD][KPAD]; // V transposed
    __shared__ __attribute__((aligned(16))) float    S[16][KPAD];  // scores / probs

    const int lane = threadIdx.x;
    const int h    = blockIdx.x;
    const int d    = blockIdx.y;
    const int head = blockIdx.z;
    const int zs = min(max(d - 2, 0), 11);
    const int ys = min(max(h - 2, 0), 11);
    const int lm = lane & 15;
    const int hi = lane >> 4;

    // ---- stage V tile transposed into LDS ----
    // t = lane + 32*i  ->  wx = lm (constant), row = hi + 2*i
    {
        const _Float16* vbase = qkv + (size_t)lm * NQKV + 384 + head * HD;
#pragma unroll
        for (int i = 0; i < 13; ++i) {
            const int row = hi + 2 * i;          // 0..24 (plus one 25 masked off)
            if (row < 25) {
                const int t  = row * 16 + lm;
                const int zi = row / 5, yi = row % 5;
                const _Float16* vr =
                    vbase + (size_t)(((zs + zi) * 16 + (ys + yi)) * 16) * NQKV;
                HFrag f0, f1;
                f0.q[0] = *(const uint4*)(vr);
                f0.q[1] = *(const uint4*)(vr + 8);
                f1.q[0] = *(const uint4*)(vr + 16);
                f1.q[1] = *(const uint4*)(vr + 24);
#pragma unroll
                for (int cc = 0; cc < 16; ++cc) Vt[cc][t] = f0.h[cc];
#pragma unroll
                for (int cc = 0; cc < 16; ++cc) Vt[16 + cc][t] = f1.h[cc];
            }
        }
    }
    if (lane < 16) {
        for (int cc = 0; cc < HD; ++cc) Vt[cc][KEYS + lane] = (_Float16)0.f;
    }

    // ---- Q fragment: A = 16 queries x 32 head-dim ----
    HFrag a;
    {
        const int vox = ((d * 16 + h) * 16) + lm;
        const _Float16* qr = qkv + (size_t)vox * NQKV + head * HD;
        const int ka = hi ? 16 : 0;
        a.q[0] = *(const uint4*)(qr + ka);
        a.q[1] = *(const uint4*)(qr + ka + 8);
    }

    // ---- scores: 25 WMMAs of Q(16x32) x K^T(32x16) ----
    const float sc = 0.17677669529663689f;       // hd^-0.5
    {
        const _Float16* kbase =
            qkv + (size_t)lm * NQKV + 192 + head * HD + (hi << 4);
#pragma unroll 5
        for (int j = 0; j < 25; ++j) {
            const int zi = j / 5, yi = j % 5;    // wave-uniform
            const _Float16* kr =
                kbase + (size_t)(((zs + zi) * 16 + (ys + yi)) * 16) * NQKV;
            HFrag b;
            b.q[0] = *(const uint4*)(kr);
            b.q[1] = *(const uint4*)(kr + 8);
            v8f cc = {};
            cc = __builtin_amdgcn_wmma_f32_16x16x32_f16(false, a.v, false, b.v,
                                                        (short)0, cc, false, false);
#pragma unroll
            for (int r = 0; r < 8; ++r) S[r + (hi << 3)][j * 16 + lm] = cc[r] * sc;
        }
    }
    __syncthreads();

    // ---- masked softmax: lanes (q, q+16) cooperate on query q ----
    {
        const int q  = lm;
        const int ws = min(max(q - 2, 0), 11);   // valid W-window [ws, ws+5)
        float mx = -3.4e38f;
        for (int t = hi; t < KEYS; t += 2) {
            const int wx = t & 15;
            if (wx >= ws && wx < ws + 5) mx = fmaxf(mx, S[q][t]);
        }
        mx = fmaxf(mx, __shfl_xor(mx, 16, 32));
        float sum = 0.f;
        for (int t = hi; t < KEYS; t += 2) {
            const int wx = t & 15;
            const bool ok = (wx >= ws) && (wx < ws + 5);
            float p = ok ? __expf(S[q][t] - mx) : 0.f;
            S[q][t] = p;
            sum += p;
        }
        sum += __shfl_xor(sum, 16, 32);
        const float inv = 1.f / sum;
        for (int t = hi; t < KEYS; t += 2) S[q][t] *= inv;
        if (hi == 0) {
            for (int t = KEYS; t < KPAD; ++t) S[q][t] = 0.f;
        }
    }
    __syncthreads();

    // ---- O = P(16x416) @ V(416x32): 13 K-chunks x 2 N-tiles ----
    v8f o0 = {}, o1 = {};
#pragma unroll
    for (int kc = 0; kc < 13; ++kc) {
        HFrag pa;
        const int kb = kc * 32 + (hi << 4);
        {
            float4 f0 = *(const float4*)(&S[lm][kb]);
            float4 f1 = *(const float4*)(&S[lm][kb + 4]);
            float4 f2 = *(const float4*)(&S[lm][kb + 8]);
            float4 f3 = *(const float4*)(&S[lm][kb + 12]);
            pa.h[0]  = (_Float16)f0.x; pa.h[1]  = (_Float16)f0.y;
            pa.h[2]  = (_Float16)f0.z; pa.h[3]  = (_Float16)f0.w;
            pa.h[4]  = (_Float16)f1.x; pa.h[5]  = (_Float16)f1.y;
            pa.h[6]  = (_Float16)f1.z; pa.h[7]  = (_Float16)f1.w;
            pa.h[8]  = (_Float16)f2.x; pa.h[9]  = (_Float16)f2.y;
            pa.h[10] = (_Float16)f2.z; pa.h[11] = (_Float16)f2.w;
            pa.h[12] = (_Float16)f3.x; pa.h[13] = (_Float16)f3.y;
            pa.h[14] = (_Float16)f3.z; pa.h[15] = (_Float16)f3.w;
        }
        const int kbb = kc * 32 + (hi << 4);
        HFrag b0, b1;
        b0.q[0] = *(const uint4*)(&Vt[lm][kbb]);
        b0.q[1] = *(const uint4*)(&Vt[lm][kbb + 8]);
        b1.q[0] = *(const uint4*)(&Vt[16 + lm][kbb]);
        b1.q[1] = *(const uint4*)(&Vt[16 + lm][kbb + 8]);
        o0 = __builtin_amdgcn_wmma_f32_16x16x32_f16(false, pa.v, false, b0.v,
                                                    (short)0, o0, false, false);
        o1 = __builtin_amdgcn_wmma_f32_16x16x32_f16(false, pa.v, false, b1.v,
                                                    (short)0, o1, false, false);
    }

    // ---- store O: obuf[vox][head*32 + hd] ----
#pragma unroll
    for (int r = 0; r < 8; ++r) {
        const int m   = r + (hi << 3);
        const int vox = ((d * 16 + h) * 16) + m;
        _Float16* orow = obuf + (size_t)vox * C_DIM + head * HD;
        orow[lm]      = (_Float16)o0[r];
        orow[16 + lm] = (_Float16)o1[r];
    }
}

// ---------------------------------------------------------------------------
// host-side orchestration
// ---------------------------------------------------------------------------
static inline int gemm_blocks64(int M, int N) { return ((M >> 6) * (N >> 4) + 7) / 8; }

extern "C" void kernel_launch(void* const* d_in, const int* in_sizes, int n_in,
                              void* d_out, int out_size, void* d_ws, size_t ws_size,
                              hipStream_t stream) {
    (void)in_sizes; (void)n_in; (void)out_size; (void)ws_size;
    const float* x      = (const float*)d_in[0];
    const float* scale1 = (const float*)d_in[1];
    const float* n1_w1  = (const float*)d_in[2];
    const float* n1_b1  = (const float*)d_in[3];
    const float* n1_w2  = (const float*)d_in[4];
    const float* n1_b2  = (const float*)d_in[5];
    const float* qkv_w  = (const float*)d_in[6];
    const float* qkv_b  = (const float*)d_in[7];
    const float* proj_w = (const float*)d_in[8];
    const float* proj_b = (const float*)d_in[9];
    const float* scale2 = (const float*)d_in[10];
    const float* n2_w1  = (const float*)d_in[11];
    const float* n2_b1  = (const float*)d_in[12];
    const float* n2_w2  = (const float*)d_in[13];
    const float* n2_b2  = (const float*)d_in[14];
    const float* mlp_w1 = (const float*)d_in[15];
    const float* mlp_b1 = (const float*)d_in[16];
    const float* mlp_w2 = (const float*)d_in[17];
    const float* mlp_b2 = (const float*)d_in[18];
    float* out = (float*)d_out;

    // workspace carve
    char* w = (char*)d_ws;
    auto carve = [&](size_t bytes) -> char* {
        char* p = w; w += (bytes + 255) & ~(size_t)255; return p;
    };
    float* stats1 = (float*)carve(C_DIM * 4);
    float* rinv1  = (float*)carve(C_DIM * 4);
    float* coef1  = (float*)carve(C_DIM * 4);
    float* stats2 = (float*)carve(C_DIM * 4);
    float* rinv2  = (float*)carve(C_DIM * 4);
    float* coef2  = (float*)carve(C_DIM * 4);
    _Float16* wq_h = (_Float16*)carve((size_t)NQKV * C_DIM * 2);
    _Float16* wp_h = (_Float16*)carve((size_t)C_DIM * C_DIM * 2);
    _Float16* w1_h = (_Float16*)carve((size_t)MLPH * C_DIM * 2);
    _Float16* w2_h = (_Float16*)carve((size_t)C_DIM * MLPH * 2);
    _Float16* hn1  = (_Float16*)carve((size_t)NVOX * C_DIM * 2);
    _Float16* qkvb = (_Float16*)carve((size_t)NVOX * NQKV * 2);
    _Float16* ob   = (_Float16*)carve((size_t)NVOX * C_DIM * 2);
    float*    x2   = (float*)   carve((size_t)NVOX * C_DIM * 4);
    _Float16* h2   = (_Float16*)carve((size_t)NVOX * C_DIM * 2);
    _Float16* m1   = (_Float16*)carve((size_t)NVOX * MLPH * 2);

    // weight conversion to f16
    {
        int n;
        n = NQKV * C_DIM;  f2h_kernel<<<(n + 255) / 256, 256, 0, stream>>>(qkv_w,  wq_h, n);
        n = C_DIM * C_DIM; f2h_kernel<<<(n + 255) / 256, 256, 0, stream>>>(proj_w, wp_h, n);
        n = MLPH * C_DIM;  f2h_kernel<<<(n + 255) / 256, 256, 0, stream>>>(mlp_w1, w1_h, n);
        n = C_DIM * MLPH;  f2h_kernel<<<(n + 255) / 256, 256, 0, stream>>>(mlp_w2, w2_h, n);
    }

    // norm 1 -> hn1
    chan_stats<<<C_DIM, 256, 0, stream>>>(x, stats1, rinv1);
    gate_kernel<<<1, HID, 0, stream>>>(stats1, rinv1, n1_w1, n1_b1, n1_w2, n1_b2,
                                       scale1, coef1);
    normalize_t<<<(C_DIM * NVOX + 255) / 256, 256, 0, stream>>>(x, coef1, hn1);

    // qkv projection (4096 x 576 x 192)
    gemm_wmma<0, C_DIM><<<gemm_blocks64(NVOX, NQKV), 256, 0, stream>>>(
        hn1, wq_h, qkv_b, NVOX, NQKV, qkvb, nullptr, nullptr);

    // neighborhood attention
    na3d_attn<<<dim3(16, 16, NH), 32, 0, stream>>>(qkvb, ob);

    // output projection + residual -> x2 (channel-major f32)
    gemm_wmma<2, C_DIM><<<gemm_blocks64(NVOX, C_DIM), 256, 0, stream>>>(
        ob, wp_h, proj_b, NVOX, C_DIM, nullptr, x2, x);

    // norm 2 -> h2
    chan_stats<<<C_DIM, 256, 0, stream>>>(x2, stats2, rinv2);
    gate_kernel<<<1, HID, 0, stream>>>(stats2, rinv2, n2_w1, n2_b1, n2_w2, n2_b2,
                                       scale2, coef2);
    normalize_t<<<(C_DIM * NVOX + 255) / 256, 256, 0, stream>>>(x2, coef2, h2);

    // MLP: 192 -> 768 (GELU) -> 192, residual -> out
    gemm_wmma<1, C_DIM><<<gemm_blocks64(NVOX, MLPH), 256, 0, stream>>>(
        h2, w1_h, mlp_b1, NVOX, MLPH, m1, nullptr, nullptr);
    gemm_wmma<2, MLPH><<<gemm_blocks64(NVOX, C_DIM), 256, 0, stream>>>(
        m1, w2_h, mlp_b2, NVOX, C_DIM, nullptr, out, x2);
}
